// GAT_72619307041134
// MI455X (gfx1250) — matrix operand order repaired and checked by reading, not compiled, not run on hardware
//
#include <hip/hip_runtime.h>
#include <hip/hip_bf16.h>

// GAT (2-layer, 8-head then 1-head) for MI455X / gfx1250.
// Dense projections use v_wmma_f32_16x16x32_f16 (f16 in, f32 accumulate).
// Edge softmax/aggregation uses per-(edge,channel) threads + f32 atomics.

typedef __attribute__((ext_vector_type(16))) _Float16 v16h;
typedef __attribute__((ext_vector_type(8)))  _Float16 v8h;
typedef __attribute__((ext_vector_type(8)))  float    v8f;

#define CDIV(a, b) (((a) + (b) - 1) / (b))

// ---------------------------------------------------------------- utilities

__global__ void fill_u32_kernel(unsigned int* __restrict__ p, unsigned int val, long n) {
  long i = (long)blockIdx.x * blockDim.x + threadIdx.x;
  if (i < n) p[i] = val;
}

__global__ void cvt_f16_kernel(const float* __restrict__ src, _Float16* __restrict__ dst, long n) {
  long i = (long)blockIdx.x * blockDim.x + threadIdx.x;
  if (i < n) dst[i] = (_Float16)src[i];
}

// Repack row-major W[K,Nout] (f32) into WMMA B-fragment order (f16):
// element ((kc*ntiles + nt)*32 + lane)*16 + j  holds  W[kc*32 + (j<8 ? half*8 + j : 16 + half*8 + j-8), nt*16 + (lane&15)]
// (dense 16-bit B 32x16 layout: lanes 0-15 hold K={0..7}U{16..23}, lanes 16-31 hold K={8..15}U{24..31},
//  2 values per VGPR, column n = lane%16)
__global__ void repack_w_kernel(const float* __restrict__ W, _Float16* __restrict__ Wp,
                                int K, int Nout) {
  int t = blockIdx.x * blockDim.x + threadIdx.x;
  int ntiles = Nout >> 4;
  int kchunks = K >> 5;
  int total = kchunks * ntiles * 32;
  if (t >= total) return;
  int lane = t & 31;
  int tmp  = t >> 5;
  int nt   = tmp % ntiles;
  int kc   = tmp / ntiles;
  int half = lane >> 4;
  int n    = nt * 16 + (lane & 15);
  _Float16* dst = Wp + (size_t)t * 16;
#pragma unroll
  for (int j = 0; j < 16; ++j) {
    int k = kc * 32 + (j < 8 ? half * 8 + j : 16 + half * 8 + (j - 8));
    dst[j] = (_Float16)W[(size_t)k * Nout + n];
  }
}

// ------------------------------------------------------------- WMMA GEMM
// C[M,Nout] = A[M,K] * B (B pre-packed).  One wave per 16x16 tile, K in 32-chunks.
__global__ void __launch_bounds__(256) wmma_gemm_kernel(
    const _Float16* __restrict__ A, const _Float16* __restrict__ Bp,
    float* __restrict__ C, int M, int K, int Nout) {
  const int lane   = threadIdx.x & 31;
  const int wid    = blockIdx.x * (blockDim.x >> 5) + (threadIdx.x >> 5);
  const int ntiles = Nout >> 4;
  const int mt     = wid / ntiles;
  const int nt     = wid - mt * ntiles;
  if (mt * 16 >= M) return;               // whole-wave uniform: EXEC stays all-ones

  const int half = lane >> 4;
  const int r    = lane & 15;
  // A fragment (16-bit A 16x32 layout): lane half 0 -> K {0..7}U{16..23},
  // half 1 -> K {8..15}U{24..31}; row = lane%16.
  const _Float16* arow = A + (size_t)(mt * 16 + r) * K + half * 8;
  const _Float16* bptr = Bp + ((size_t)nt * 32 + lane) * 16;
  const size_t bstride = (size_t)ntiles * 32 * 16;   // per 32-K chunk

  v8f acc = {};
  const int kchunks = K >> 5;
  for (int kc = 0; kc < kchunks; ++kc) {
    v8h alo = *(const v8h*)(arow + kc * 32);        // 16B load
    v8h ahi = *(const v8h*)(arow + kc * 32 + 16);   // 16B load
    v16h a  = __builtin_shufflevector(alo, ahi, 0, 1, 2, 3, 4, 5, 6, 7,
                                      8, 9, 10, 11, 12, 13, 14, 15);
    v16h b  = *(const v16h*)(bptr + (size_t)kc * bstride);  // 32B contiguous
    acc = __builtin_amdgcn_wmma_f32_16x16x32_f16(
        /*neg_a=*/false, a, /*neg_b=*/false, b,
        /*c_mod=*/(short)0, acc, /*reuse_a=*/false, /*reuse_b=*/false);
  }
  // D layout: VGPR v -> row mt*16 + v + 8*half, col nt*16 + (lane&15)
  float* crow = C + (size_t)(mt * 16 + half * 8) * Nout + nt * 16 + r;
#pragma unroll
  for (int v = 0; v < 8; ++v) crow[(size_t)v * Nout] = acc[v];
}

// -------------------------------------------------------- attention logits
// als[n,h] = sum_c h[n,h,c]*a_src[h,c]   (and ald likewise)
__global__ void logits_kernel(const float* __restrict__ h,
                              const float* __restrict__ a_src,
                              const float* __restrict__ a_dst,
                              float* __restrict__ als, float* __restrict__ ald,
                              int N, int H, int CH) {
  long t = (long)blockIdx.x * blockDim.x + threadIdx.x;
  if (t >= (long)N * H) return;
  int n = (int)(t / H), hh = (int)(t % H);
  const float* hp = h + (size_t)n * H * CH + (size_t)hh * CH;
  const float* as = a_src + (size_t)hh * CH;
  const float* ad = a_dst + (size_t)hh * CH;
  float s = 0.f, d = 0.f;
  for (int c = 0; c < CH; ++c) {
    float v = hp[c];
    s += v * as[c];
    d += v * ad[c];
  }
  als[t] = s;
  ald[t] = d;
}

// ----------------------------------------------------------- edge passes

__device__ inline void atomicMaxFloat(float* addr, float val) {
  // float-ordered max via integer atomics (init value must be -inf)
  if (val >= 0.f) atomicMax((int*)addr, __float_as_int(val));
  else            atomicMin((unsigned int*)addr, __float_as_uint(val));
}

__device__ inline void edge_ends(const int* ei, int E, int k, int& s, int& d) {
  if (k < E) { s = ei[k]; d = ei[E + k]; }
  else       { s = k - E; d = k - E; }      // self loop
}

template <int H>
__global__ void edge_max_kernel(const int* __restrict__ ei, int E, int Etot,
                                const float* __restrict__ als,
                                const float* __restrict__ ald,
                                float* __restrict__ emax) {
  long t = (long)blockIdx.x * blockDim.x + threadIdx.x;
  if (t >= (long)Etot * H) return;
  int k = (int)(t / H), h = (int)(t % H);
  int s, d;
  edge_ends(ei, E, k, s, d);
  float e = als[(size_t)s * H + h] + ald[(size_t)d * H + h];
  e = e > 0.f ? e : 0.2f * e;               // leaky relu
  atomicMaxFloat(emax + (size_t)d * H + h, e);
}

template <int H>
__global__ void edge_expsum_kernel(const int* __restrict__ ei, int E, int Etot,
                                   const float* __restrict__ als,
                                   const float* __restrict__ ald,
                                   const float* __restrict__ emax,
                                   float* __restrict__ den,
                                   float* __restrict__ ee) {
  long t = (long)blockIdx.x * blockDim.x + threadIdx.x;
  if (t >= (long)Etot * H) return;
  int k = (int)(t / H), h = (int)(t % H);
  int s, d;
  edge_ends(ei, E, k, s, d);
  float e = als[(size_t)s * H + h] + ald[(size_t)d * H + h];
  e = e > 0.f ? e : 0.2f * e;
  float x = __expf(e - emax[(size_t)d * H + h]);
  ee[t] = x;
  atomicAdd(den + (size_t)d * H + h, x);
}

template <int H, int CTOT>
__global__ void edge_agg_kernel(const int* __restrict__ ei, int E, int Etot,
                                const float* __restrict__ hfeat,
                                const float* __restrict__ ee,
                                const float* __restrict__ den,
                                float* __restrict__ agg) {
  long t = (long)blockIdx.x * blockDim.x + threadIdx.x;
  if (t >= (long)Etot * CTOT) return;
  int k = (int)(t / CTOT), c = (int)(t % CTOT);
  constexpr int CH = CTOT / H;
  int h = c / CH;
  int s, d;
  edge_ends(ei, E, k, s, d);
  float alpha = ee[(size_t)k * H + h] / (den[(size_t)d * H + h] + 1e-16f);
  atomicAdd(agg + (size_t)d * CTOT + c, hfeat[(size_t)s * CTOT + c] * alpha);
}

// ------------------------------------------------------------- epilogues

__global__ void relu_bias_f16_kernel(const float* __restrict__ agg,
                                     const float* __restrict__ bias,
                                     _Float16* __restrict__ out, long n, int C) {
  long t = (long)blockIdx.x * blockDim.x + threadIdx.x;
  if (t >= n) return;
  int c = (int)(t % C);
  float v = agg[t] + bias[c];
  out[t] = (_Float16)(v > 0.f ? v : 0.f);
}

__global__ void bias_out_kernel(const float* __restrict__ agg,
                                const float* __restrict__ bias,
                                float* __restrict__ out, long n, int C) {
  long t = (long)blockIdx.x * blockDim.x + threadIdx.x;
  if (t >= n) return;
  int c = (int)(t % C);
  out[t] = agg[t] + bias[c];
}

// ------------------------------------------------------------------ launch

extern "C" void kernel_launch(void* const* d_in, const int* in_sizes, int n_in,
                              void* d_out, int out_size, void* d_ws, size_t ws_size,
                              hipStream_t stream) {
  const float* x   = (const float*)d_in[0];
  const int*   ei  = (const int*)d_in[1];
  const float* W1  = (const float*)d_in[2];
  const float* a1s = (const float*)d_in[3];
  const float* a1d = (const float*)d_in[4];
  const float* b1  = (const float*)d_in[5];
  const float* W2  = (const float*)d_in[6];
  const float* a2s = (const float*)d_in[7];
  const float* a2d = (const float*)d_in[8];
  const float* b2  = (const float*)d_in[9];
  (void)n_in; (void)out_size; (void)ws_size;

  const int N    = in_sizes[0] / 128;   // 50000
  const int E    = in_sizes[1] / 2;     // 800000
  const int Etot = E + N;               // +self loops

  // ---- carve workspace (layer 2 reuses layer-1 regions once they're dead)
  char* base = (char*)d_ws;
  size_t off = 0;
  auto carve = [&](size_t bytes) -> void* {
    void* p = base + off;
    off += (bytes + 255) & ~(size_t)255;
    return p;
  };
  _Float16* xh    = (_Float16*)carve((size_t)N * 128 * 2);
  _Float16* W1p   = (_Float16*)carve((size_t)128 * 128 * 2);
  _Float16* W2p   = (_Float16*)carve((size_t)128 * 64 * 2);
  float*    h1    = (float*)carve((size_t)N * 128 * 4);
  float*    al1s  = (float*)carve((size_t)N * 8 * 4);
  float*    al1d  = (float*)carve((size_t)N * 8 * 4);
  float*    emax1 = (float*)carve((size_t)N * 8 * 4);
  float*    den1  = (float*)carve((size_t)N * 8 * 4);
  float*    ee1   = (float*)carve((size_t)Etot * 8 * 4);
  float*    agg1  = (float*)carve((size_t)N * 128 * 4);
  _Float16* h1rh  = (_Float16*)carve((size_t)N * 128 * 2);
  // layer-2 aliases (regions are >= required sizes)
  float* h2 = h1;     float* al2s = al1s;  float* al2d = al1d;
  float* emax2 = emax1; float* den2 = den1; float* ee2 = ee1; float* agg2 = agg1;

  const int BS = 256;
  const int mtiles = CDIV(N, 16);

  // ---- layer 1 ----
  long n_xh = (long)N * 128;
  cvt_f16_kernel<<<(unsigned)CDIV(n_xh, BS), BS, 0, stream>>>(x, xh, n_xh);
  repack_w_kernel<<<(unsigned)CDIV(4 * 8 * 32, BS), BS, 0, stream>>>(W1, W1p, 128, 128);
  repack_w_kernel<<<(unsigned)CDIV(4 * 4 * 32, BS), BS, 0, stream>>>(W2, W2p, 128, 64);

  long waves1 = (long)mtiles * 8;
  wmma_gemm_kernel<<<(unsigned)CDIV(waves1 * 32, BS), BS, 0, stream>>>(xh, W1p, h1, N, 128, 128);

  logits_kernel<<<(unsigned)CDIV((long)N * 8, BS), BS, 0, stream>>>(h1, a1s, a1d, al1s, al1d, N, 8, 16);

  fill_u32_kernel<<<(unsigned)CDIV((long)N * 8, BS), BS, 0, stream>>>((unsigned*)emax1, 0xFF800000u, (long)N * 8);
  fill_u32_kernel<<<(unsigned)CDIV((long)N * 8, BS), BS, 0, stream>>>((unsigned*)den1, 0u, (long)N * 8);
  fill_u32_kernel<<<(unsigned)CDIV((long)N * 128, BS), BS, 0, stream>>>((unsigned*)agg1, 0u, (long)N * 128);

  edge_max_kernel<8><<<(unsigned)CDIV((long)Etot * 8, BS), BS, 0, stream>>>(ei, E, Etot, al1s, al1d, emax1);
  edge_expsum_kernel<8><<<(unsigned)CDIV((long)Etot * 8, BS), BS, 0, stream>>>(ei, E, Etot, al1s, al1d, emax1, den1, ee1);
  edge_agg_kernel<8, 128><<<(unsigned)CDIV((long)Etot * 128, BS), BS, 0, stream>>>(ei, E, Etot, h1, ee1, den1, agg1);

  relu_bias_f16_kernel<<<(unsigned)CDIV((long)N * 128, BS), BS, 0, stream>>>(agg1, b1, h1rh, (long)N * 128, 128);

  // ---- layer 2 ----
  long waves2 = (long)mtiles * 4;
  wmma_gemm_kernel<<<(unsigned)CDIV(waves2 * 32, BS), BS, 0, stream>>>(h1rh, W2p, h2, N, 128, 64);

  logits_kernel<<<(unsigned)CDIV((long)N, BS), BS, 0, stream>>>(h2, a2s, a2d, al2s, al2d, N, 1, 64);

  fill_u32_kernel<<<(unsigned)CDIV((long)N, BS), BS, 0, stream>>>((unsigned*)emax2, 0xFF800000u, (long)N);
  fill_u32_kernel<<<(unsigned)CDIV((long)N, BS), BS, 0, stream>>>((unsigned*)den2, 0u, (long)N);
  fill_u32_kernel<<<(unsigned)CDIV((long)N * 64, BS), BS, 0, stream>>>((unsigned*)agg2, 0u, (long)N * 64);

  edge_max_kernel<1><<<(unsigned)CDIV((long)Etot, BS), BS, 0, stream>>>(ei, E, Etot, al2s, al2d, emax2);
  edge_expsum_kernel<1><<<(unsigned)CDIV((long)Etot, BS), BS, 0, stream>>>(ei, E, Etot, al2s, al2d, emax2, den2, ee2);
  edge_agg_kernel<1, 64><<<(unsigned)CDIV((long)Etot * 64, BS), BS, 0, stream>>>(ei, E, Etot, h2, ee2, den2, agg2);

  bias_out_kernel<<<(unsigned)CDIV((long)N * 64, BS), BS, 0, stream>>>(agg2, b2, (float*)d_out, (long)N * 64, 64);
}
